// Qwen3MoeAttention_63067299775015
// MI455X (gfx1250) — compile-verified, matrix operand order checked by reading, not run on hardware
//
#include <hip/hip_runtime.h>
#include <hip/hip_bf16.h>

// ---------------------------------------------------------------------------
// Qwen3-MoE-style attention block for MI455X (gfx1250, wave32, WMMA).
// All GEMM inner products run on v_wmma_f32_16x16x32_bf16 (fp32 accumulate).
// ---------------------------------------------------------------------------

#define B_   2
#define S_   2048
#define HID_ 2048
#define H_   16
#define KV_  4
#define D_   128
#define E_   32
#define DE_  (D_ + E_)          // 160
#define EPS_ 1e-6f
// (D+E)^-0.5 = 160^-0.5
#define SCALING_ 0.07905694150420949f

typedef __attribute__((ext_vector_type(16))) __bf16         v16bf;
typedef __attribute__((ext_vector_type(8)))  float          v8f;
typedef __attribute__((ext_vector_type(8)))  unsigned short us8;
typedef __attribute__((ext_vector_type(4)))  unsigned short us4;

union BF16Frag {
    v16bf bf;
    us8   u8[2];
};

// float -> bf16 bits, round-to-nearest-even
__device__ __forceinline__ unsigned short f2bf(float f) {
    unsigned int u = __float_as_uint(f);
    unsigned int r = u + 0x7FFFu + ((u >> 16) & 1u);
    return (unsigned short)(r >> 16);
}

// ---------------------------------------------------------------------------
// GEMM: C[M,N] (f32) = A[M,K] (f32) * B[N,K]^T (f32), inner products in bf16.
// Block tile 128x128, K-step 32. 8 waves; each wave computes 32x64 (2x4 frags).
// M,N multiples of 128; K multiple of 32 (true for all uses here).
// ---------------------------------------------------------------------------
__global__ __launch_bounds__(256) void gemm_bf16wmma(
    const float* __restrict__ A, const float* __restrict__ Bw,
    float* __restrict__ C, int M, int N, int K)
{
    constexpr int TM = 128, TN = 128, TK = 32, LS = 48; // LS: LDS row stride (halves)
    __shared__ __align__(16) unsigned short As[TM * LS];
    __shared__ __align__(16) unsigned short Bs[TN * LS];

    const int tid  = threadIdx.x;
    const int lane = tid & 31;
    const int wid  = tid >> 5;
    const int wm   = wid & 3;     // 4 waves along M
    const int wn   = wid >> 2;    // 2 waves along N
    const int l15  = lane & 15;
    const int half = lane >> 4;
    const int hk   = half ? 8 : 0;

    const size_t bm = (size_t)blockIdx.y * TM;
    const size_t bn = (size_t)blockIdx.x * TN;

    v8f acc[2][4] = {};

    for (int k0 = 0; k0 < K; k0 += TK) {
        __syncthreads();
        // Stage 128x32 f32 -> bf16 tiles of A and B into LDS.
        // 1024 float4-chunks per tile, 4 per thread.
#pragma unroll
        for (int i = 0; i < 4; ++i) {
            const int idx = tid + i * 256;
            const int row = idx >> 3;
            const int c4  = idx & 7;
            const float4 a = *reinterpret_cast<const float4*>(
                A + (bm + row) * (size_t)K + k0 + c4 * 4);
            us4 pa = { f2bf(a.x), f2bf(a.y), f2bf(a.z), f2bf(a.w) };
            *reinterpret_cast<us4*>(&As[row * LS + c4 * 4]) = pa;
            const float4 b = *reinterpret_cast<const float4*>(
                Bw + (bn + row) * (size_t)K + k0 + c4 * 4);
            us4 pb = { f2bf(b.x), f2bf(b.y), f2bf(b.z), f2bf(b.w) };
            *reinterpret_cast<us4*>(&Bs[row * LS + c4 * 4]) = pb;
        }
        __syncthreads();

        // Fragment loads per ISA 16x32 bf16 layout:
        // lane row = l15; K chunks {hk..hk+7, hk+16..hk+23}.
        BF16Frag af[2], bf[4];
#pragma unroll
        for (int i = 0; i < 2; ++i) {
            const int r = wm * 32 + i * 16 + l15;
            af[i].u8[0] = *reinterpret_cast<const us8*>(&As[r * LS + hk]);
            af[i].u8[1] = *reinterpret_cast<const us8*>(&As[r * LS + hk + 16]);
        }
#pragma unroll
        for (int j = 0; j < 4; ++j) {
            const int r = wn * 64 + j * 16 + l15;
            bf[j].u8[0] = *reinterpret_cast<const us8*>(&Bs[r * LS + hk]);
            bf[j].u8[1] = *reinterpret_cast<const us8*>(&Bs[r * LS + hk + 16]);
        }
#pragma unroll
        for (int i = 0; i < 2; ++i)
#pragma unroll
            for (int j = 0; j < 4; ++j)
                acc[i][j] = __builtin_amdgcn_wmma_f32_16x16x32_bf16(
                    false, af[i].bf, false, bf[j].bf, (short)0, acc[i][j],
                    false, false);
    }

    // C layout: VGPR r, lanes0-15 -> M=r, lanes16-31 -> M=8+r; N = lane&15.
#pragma unroll
    for (int i = 0; i < 2; ++i)
#pragma unroll
        for (int j = 0; j < 4; ++j)
#pragma unroll
            for (int r = 0; r < 8; ++r) {
                const size_t m = bm + wm * 32 + i * 16 + 8 * half + r;
                const size_t n = bn + wn * 64 + j * 16 + l15;
                C[m * (size_t)N + n] = acc[i][j][r];
            }
}

// ---------------------------------------------------------------------------
// Per-token fuse: RMSNorm(q,k per head), behavior-embedding gather + RMSNorm,
// concat to bf16 q_cat[B,H,S,160] / k_cat[B,KV,S,160]; V -> bf16 Vt[B,KV,D,S].
// One block per (b,s); 8 waves with disjoint jobs.
// ---------------------------------------------------------------------------
__global__ __launch_bounds__(256) void fuse_norm_concat(
    const float* __restrict__ qbuf,   // [B*S, H*D]
    const float* __restrict__ kbuf,   // [B*S, KV*D]
    const float* __restrict__ vbuf,   // [B*S, KV*D]
    const int*   __restrict__ bidx,   // [B*S]
    const float* __restrict__ qnw, const float* __restrict__ knw,
    const float* __restrict__ qbeT,   // [NB, H*E]
    const float* __restrict__ kbeT,   // [NB, KV*E]
    const float* __restrict__ qbnw, const float* __restrict__ kbnw,
    unsigned short* __restrict__ qcat,
    unsigned short* __restrict__ kcat,
    unsigned short* __restrict__ vt)
{
    const int row  = blockIdx.x;            // b*S + s
    const int b    = row >> 11;             // /S_
    const int s    = row & (S_ - 1);
    const int tid  = threadIdx.x;
    const int lane = tid & 31;
    const int wid  = tid >> 5;
    const int beh  = bidx[row];

    if (wid < 4) {                          // q: 16 heads, 4 per wave
#pragma unroll
        for (int hh = 0; hh < 4; ++hh) {
            const int h = wid * 4 + hh;
            const float4 x = *reinterpret_cast<const float4*>(
                qbuf + ((size_t)row * H_ + h) * D_ + lane * 4);
            float ss = x.x * x.x + x.y * x.y + x.z * x.z + x.w * x.w;
            for (int m = 16; m >= 1; m >>= 1) ss += __shfl_xor(ss, m, 32);
            const float inv = rsqrtf(ss * (1.f / D_) + EPS_);
            unsigned short* dst =
                qcat + ((size_t)(b * H_ + h) * S_ + s) * DE_ + lane * 4;
            dst[0] = f2bf(x.x * inv * qnw[lane * 4 + 0]);
            dst[1] = f2bf(x.y * inv * qnw[lane * 4 + 1]);
            dst[2] = f2bf(x.z * inv * qnw[lane * 4 + 2]);
            dst[3] = f2bf(x.w * inv * qnw[lane * 4 + 3]);
        }
    } else if (wid == 4) {                  // k: 4 kv heads
#pragma unroll
        for (int hh = 0; hh < KV_; ++hh) {
            const float4 x = *reinterpret_cast<const float4*>(
                kbuf + ((size_t)row * KV_ + hh) * D_ + lane * 4);
            float ss = x.x * x.x + x.y * x.y + x.z * x.z + x.w * x.w;
            for (int m = 16; m >= 1; m >>= 1) ss += __shfl_xor(ss, m, 32);
            const float inv = rsqrtf(ss * (1.f / D_) + EPS_);
            unsigned short* dst =
                kcat + ((size_t)(b * KV_ + hh) * S_ + s) * DE_ + lane * 4;
            dst[0] = f2bf(x.x * inv * knw[lane * 4 + 0]);
            dst[1] = f2bf(x.y * inv * knw[lane * 4 + 1]);
            dst[2] = f2bf(x.z * inv * knw[lane * 4 + 2]);
            dst[3] = f2bf(x.w * inv * knw[lane * 4 + 3]);
        }
    } else if (wid == 5) {                  // v transpose -> bf16 Vt[B,KV,D,S]
#pragma unroll
        for (int i = 0; i < 16; ++i) {
            const int e  = lane * 16 + i;
            const int kv = e >> 7;
            const int d  = e & 127;
            const float val = vbuf[(size_t)row * (KV_ * D_) + e];
            vt[((size_t)(b * KV_ + kv) * D_ + d) * S_ + s] = f2bf(val);
        }
    } else if (wid == 6) {                  // q behavior embedding (E=32: 1 lane/elem)
        const float* t = qbeT + (size_t)beh * H_ * E_;
        for (int h = 0; h < H_; ++h) {
            const float x = t[h * E_ + lane];
            float ss = x * x;
            for (int m = 16; m >= 1; m >>= 1) ss += __shfl_xor(ss, m, 32);
            const float inv = rsqrtf(ss * (1.f / E_) + EPS_);
            qcat[((size_t)(b * H_ + h) * S_ + s) * DE_ + D_ + lane] =
                f2bf(x * inv * qbnw[lane]);
        }
    } else {                                // k behavior embedding
        const float* t = kbeT + (size_t)beh * KV_ * E_;
        for (int h = 0; h < KV_; ++h) {
            const float x = t[h * E_ + lane];
            float ss = x * x;
            for (int m = 16; m >= 1; m >>= 1) ss += __shfl_xor(ss, m, 32);
            const float inv = rsqrtf(ss * (1.f / E_) + EPS_);
            kcat[((size_t)(b * KV_ + h) * S_ + s) * DE_ + D_ + lane] =
                f2bf(x * inv * kbnw[lane]);
        }
    }
}

// ---------------------------------------------------------------------------
// Flash attention: one wave per 16-row Q tile; online softmax; keys in chunks
// of 32 (two 16x16 score tiles -> one K=32 PV WMMA set). P goes through LDS
// to convert from WMMA C-layout to A-fragment layout.
// ---------------------------------------------------------------------------
__global__ __launch_bounds__(128) void flash_attn(
    const unsigned short* __restrict__ qcat,   // [B,H,S,160] bf16
    const unsigned short* __restrict__ kcat,   // [B,KV,S,160] bf16
    const unsigned short* __restrict__ vt,     // [B,KV,D,S] bf16
    float* __restrict__ attn)                  // [B,S,H,D] f32
{
    __shared__ __align__(16) unsigned short pls[4 * 16 * 48];

    const int tid  = threadIdx.x;
    const int lane = tid & 31;
    const int wid  = tid >> 5;
    const int w    = blockIdx.x * 4 + wid;  // global wave id, 4096 total
    const int qt   = w & 127;               // S/16 = 128 q tiles
    const int bh   = w >> 7;
    const int h    = bh & (H_ - 1);
    const int b    = bh >> 4;
    const int kvh  = h >> 2;                // H/KV = 4
    const int l15  = lane & 15;
    const int half = lane >> 4;
    const int hk   = half ? 8 : 0;

    const unsigned short* qp = qcat + (size_t)(b * H_ + h) * S_ * DE_;
    const unsigned short* kp = kcat + (size_t)(b * KV_ + kvh) * S_ * DE_;
    const unsigned short* vp = vt + (size_t)(b * KV_ + kvh) * D_ * S_;

    // Load 5 Q A-fragments (16 rows x 160 features)
    BF16Frag qf[5];
    {
        const int qrow = qt * 16 + l15;
#pragma unroll
        for (int f = 0; f < 5; ++f) {
            const unsigned short* p = qp + (size_t)qrow * DE_ + f * 32 + hk;
            qf[f].u8[0] = *reinterpret_cast<const us8*>(p);
            qf[f].u8[1] = *reinterpret_cast<const us8*>(p + 16);
        }
    }

    v8f acc[8] = {};
    float mrow[8], lrow[8];
#pragma unroll
    for (int r = 0; r < 8; ++r) { mrow[r] = -3.0e38f; lrow[r] = 0.f; }

    unsigned short* pw = &pls[wid * 16 * 48];
    const int nchunk = (qt * 16 + 16 + 31) >> 5;

    for (int c = 0; c < nchunk; ++c) {
        const int kb = c * 32;

        // scores: Q(16x160) @ K^T(160x32) via 2 column tiles x 5 chained WMMA
        v8f s0 = {}, s1 = {};
#pragma unroll
        for (int f = 0; f < 5; ++f) {
            BF16Frag kf;
            const unsigned short* p =
                kp + (size_t)(kb + l15) * DE_ + f * 32 + hk;
            kf.u8[0] = *reinterpret_cast<const us8*>(p);
            kf.u8[1] = *reinterpret_cast<const us8*>(p + 16);
            s0 = __builtin_amdgcn_wmma_f32_16x16x32_bf16(
                false, qf[f].bf, false, kf.bf, (short)0, s0, false, false);
        }
#pragma unroll
        for (int f = 0; f < 5; ++f) {
            BF16Frag kf;
            const unsigned short* p =
                kp + (size_t)(kb + 16 + l15) * DE_ + f * 32 + hk;
            kf.u8[0] = *reinterpret_cast<const us8*>(p);
            kf.u8[1] = *reinterpret_cast<const us8*>(p + 16);
            s1 = __builtin_amdgcn_wmma_f32_16x16x32_bf16(
                false, qf[f].bf, false, kf.bf, (short)0, s1, false, false);
        }

        // causal mask + online softmax (C layout: row = 8*half + r, col = l15)
        float p0[8], p1[8], esc[8];
#pragma unroll
        for (int r = 0; r < 8; ++r) {
            const int rowg = qt * 16 + 8 * half + r;
            const int c0 = kb + l15, c1 = kb + 16 + l15;
            const float v0 = (c0 <= rowg) ? s0[r] * SCALING_ : -3.0e38f;
            const float v1 = (c1 <= rowg) ? s1[r] * SCALING_ : -3.0e38f;
            float m2 = fmaxf(v0, v1);
            for (int msk = 1; msk < 16; msk <<= 1)
                m2 = fmaxf(m2, __shfl_xor(m2, msk, 32));
            const float nm = fmaxf(mrow[r], m2);
            const float e0 = __expf(v0 - nm);
            const float e1 = __expf(v1 - nm);
            float sum = e0 + e1;
            for (int msk = 1; msk < 16; msk <<= 1)
                sum += __shfl_xor(sum, msk, 32);
            const float sc = __expf(mrow[r] - nm);
            lrow[r] = lrow[r] * sc + sum;
            mrow[r] = nm;
            esc[r]  = sc;
            p0[r]   = e0;
            p1[r]   = e1;
        }
#pragma unroll
        for (int t = 0; t < 8; ++t)
#pragma unroll
            for (int r = 0; r < 8; ++r) acc[t][r] *= esc[r];

        // P (C layout) -> LDS -> A fragment layout (DS ops in-order per wave)
#pragma unroll
        for (int r = 0; r < 8; ++r) {
            const int m = 8 * half + r;
            pw[m * 48 + l15]      = f2bf(p0[r]);
            pw[m * 48 + 16 + l15] = f2bf(p1[r]);
        }
        BF16Frag pf;
        pf.u8[0] = *reinterpret_cast<const us8*>(&pw[l15 * 48 + hk]);
        pf.u8[1] = *reinterpret_cast<const us8*>(&pw[l15 * 48 + hk + 16]);

        // O += P(16x32) @ V(32x128): 8 column tiles of 16
#pragma unroll
        for (int t = 0; t < 8; ++t) {
            BF16Frag vf;
            const unsigned short* p =
                vp + (size_t)(t * 16 + l15) * S_ + kb + hk;
            vf.u8[0] = *reinterpret_cast<const us8*>(p);
            vf.u8[1] = *reinterpret_cast<const us8*>(p + 16);
            acc[t] = __builtin_amdgcn_wmma_f32_16x16x32_bf16(
                false, pf.bf, false, vf.bf, (short)0, acc[t], false, false);
        }
    }

    // epilogue: divide by row sums, write [b, s, h, d] f32 for o_proj
#pragma unroll
    for (int r = 0; r < 8; ++r) {
        const int rowg  = qt * 16 + 8 * half + r;
        const float inv = 1.f / lrow[r];
#pragma unroll
        for (int t = 0; t < 8; ++t)
            attn[((size_t)(b * S_ + rowg) * H_ + h) * D_ + t * 16 + l15] =
                acc[t][r] * inv;
    }
}

// ---------------------------------------------------------------------------
extern "C" void kernel_launch(void* const* d_in, const int* in_sizes, int n_in,
                              void* d_out, int out_size, void* d_ws,
                              size_t ws_size, hipStream_t stream)
{
    (void)in_sizes; (void)n_in; (void)out_size; (void)ws_size;
    const float* hidden = (const float*)d_in[0];
    const int*   bidx   = (const int*)d_in[1];
    const float* qw     = (const float*)d_in[2];
    const float* kw     = (const float*)d_in[3];
    const float* vw     = (const float*)d_in[4];
    const float* ow     = (const float*)d_in[5];
    const float* qnw    = (const float*)d_in[6];
    const float* knw    = (const float*)d_in[7];
    const float* qbeT   = (const float*)d_in[8];
    const float* kbeT   = (const float*)d_in[9];
    const float* qbnw   = (const float*)d_in[10];
    const float* kbnw   = (const float*)d_in[11];
    float* out = (float*)d_out;

    char*  ws  = (char*)d_ws;
    size_t off = 0;
    auto alloc = [&](size_t bytes) -> void* {
        off = (off + 255) & ~(size_t)255;
        void* p = ws + off;
        off += bytes;
        return p;
    };
    float* qbuf = (float*)alloc(sizeof(float) * (size_t)B_ * S_ * H_ * D_);
    float* kbuf = (float*)alloc(sizeof(float) * (size_t)B_ * S_ * KV_ * D_);
    float* vbuf = (float*)alloc(sizeof(float) * (size_t)B_ * S_ * KV_ * D_);
    float* attn = (float*)alloc(sizeof(float) * (size_t)B_ * S_ * H_ * D_);
    unsigned short* qcat =
        (unsigned short*)alloc(2ull * B_ * H_ * S_ * DE_);
    unsigned short* kcat =
        (unsigned short*)alloc(2ull * B_ * KV_ * S_ * DE_);
    unsigned short* vt =
        (unsigned short*)alloc(2ull * B_ * KV_ * D_ * S_);

    const int M = B_ * S_;

    // QKV projections
    gemm_bf16wmma<<<dim3((H_ * D_) / 128, M / 128), 256, 0, stream>>>(
        hidden, qw, qbuf, M, H_ * D_, HID_);
    gemm_bf16wmma<<<dim3((KV_ * D_) / 128, M / 128), 256, 0, stream>>>(
        hidden, kw, kbuf, M, KV_ * D_, HID_);
    gemm_bf16wmma<<<dim3((KV_ * D_) / 128, M / 128), 256, 0, stream>>>(
        hidden, vw, vbuf, M, KV_ * D_, HID_);

    // RMSNorms + behavior-embedding concat + V transpose (all -> bf16)
    fuse_norm_concat<<<M, 256, 0, stream>>>(
        qbuf, kbuf, vbuf, bidx, qnw, knw, qbeT, kbeT, qbnw, kbnw,
        qcat, kcat, vt);

    // causal GQA flash attention: 4096 waves, 4 per block
    flash_attn<<<(B_ * H_ * (S_ / 16)) / 4, 128, 0, stream>>>(
        qcat, kcat, vt, attn);

    // output projection -> d_out
    gemm_bf16wmma<<<dim3(HID_ / 128, M / 128), 256, 0, stream>>>(
        attn, ow, out, M, HID_, H_ * D_);
}
